// Decoder_27367531611012
// MI455X (gfx1250) — matrix operand order repaired and checked by reading, not compiled, not run on hardware
//
#include <hip/hip_runtime.h>
#include <hip/hip_bf16.h>
#include <math.h>

#define B_    512
#define S_    128
#define E_    128
#define H_    128
#define G4_   512   // 4*H
#define KCAT_ 256   // E + H concatenated input for gates GEMM
#define NEGV  (-1e30f)

typedef __attribute__((ext_vector_type(16))) __bf16        v16bf;
typedef __attribute__((ext_vector_type(8)))  float         v8f;
typedef __attribute__((ext_vector_type(2)))  unsigned int  u32x2;
typedef __attribute__((ext_vector_type(4)))  unsigned int  u32x4;
typedef __attribute__((ext_vector_type(8)))  unsigned int  u32x8;

// ---------- bf16 helpers (bit-level, round-to-nearest-even) ----------
__device__ __forceinline__ unsigned short f2bf(float f) {
  unsigned int u = __builtin_bit_cast(unsigned int, f);
  unsigned int r = u + 0x7FFFu + ((u >> 16) & 1u);
  return (unsigned short)(r >> 16);
}
__device__ __forceinline__ float bf2f(unsigned short h) {
  unsigned int u = ((unsigned int)h) << 16;
  return __builtin_bit_cast(float, u);
}
__device__ __forceinline__ float bf2f_lo(unsigned int u) {
  return __builtin_bit_cast(float, u << 16);
}
__device__ __forceinline__ float bf2f_hi(unsigned int u) {
  return __builtin_bit_cast(float, u & 0xFFFF0000u);
}

// ---------- generic f32 -> bf16 convert with strides ----------
__global__ void k_cvt(const float* __restrict__ src, unsigned short* __restrict__ dst,
                      int rows, int cols, int lds, int ldd) {
  int i = blockIdx.x * blockDim.x + threadIdx.x;
  if (i >= rows * cols) return;
  int r = i / cols, c = i - r * cols;
  dst[(size_t)r * ldd + c] = f2bf(src[(size_t)r * lds + c]);
}

__global__ void k_addbias(const float* __restrict__ a, const float* __restrict__ b,
                          float* __restrict__ out, int n) {
  int i = blockIdx.x * blockDim.x + threadIdx.x;
  if (i < n) out[i] = a[i] + b[i];
}

// ---------- pack a (KxN) f32 weight into WMMA-B fragment-native bf16 layout ----
// Layout: Wp[((ntile*(K/32) + kc)*32 + lane)*16 + e], where for lane = 16*half+l16
// element e corresponds to B[k = kc*32 + 16*(e/8) + 8*half + e%8][n = ntile*16 + l16].
// Split-K concat supported: rows [0,splitK) from W0, rows [splitK,K) from W1.
__global__ void k_packW(const float* __restrict__ W0, const float* __restrict__ W1,
                        int splitK, int ldw, unsigned short* __restrict__ Wp,
                        int N, int K) {
  int i = blockIdx.x * blockDim.x + threadIdx.x;
  int total = (N >> 4) * (K >> 5) * 32 * 16;
  if (i >= total) return;
  int e     = i & 15;
  int lane  = (i >> 4) & 31;
  int kc    = (i >> 9) % (K >> 5);
  int ntile = (i >> 9) / (K >> 5);
  int half = lane >> 4, l16 = lane & 15;
  int k = kc * 32 + 16 * (e >> 3) + 8 * half + (e & 7);
  int n = ntile * 16 + l16;
  float v = (k < splitK) ? W0[(size_t)k * ldw + n]
                         : W1[(size_t)(k - splitK) * ldw + n];
  Wp[i] = f2bf(v);
}

// ---------- WMMA bf16 GEMM: C(MxN) = A(MxK) * W(KxN) + bias(N) ----------
// One wave32 per 16x16 output tile. A fragment = two contiguous b128 loads
// per lane (two 8-wide K runs: 8*half+0..7 and 16+8*half+0..7); B fragment = one
// contiguous 32B load from the pre-packed weight buffer.
template<bool OUT_BF16>
__global__ void k_gemm(const unsigned short* __restrict__ A, int lda,
                       const unsigned short* __restrict__ Wp,
                       const float* __restrict__ bias,
                       void* __restrict__ C, int ldc,
                       int M, int N, int K) {
  int wave  = blockIdx.x * (blockDim.x >> 5) + (threadIdx.x >> 5);
  int lane  = threadIdx.x & 31;
  int tilesN = N >> 4;
  int tiles  = (M >> 4) * tilesN;
  if (wave >= tiles) return;
  int tm  = (wave / tilesN) << 4;
  int tnb = wave % tilesN;
  int half = lane >> 4;
  int l16  = lane & 15;
  int kchunks = K >> 5;

  v8f acc = {};
  const unsigned short* Arow = A + (size_t)(tm + l16) * lda + 8 * half;
  const unsigned short* Wl   = Wp + ((size_t)tnb * kchunks * 32 + lane) * 16;

  for (int kc = 0; kc < kchunks; ++kc) {
    u32x4 alo = *(const u32x4*)(Arow + kc * 32);        // K = 8*half + 0..7
    u32x4 ahi = *(const u32x4*)(Arow + kc * 32 + 16);   // K = 16 + 8*half + 0..7
    u32x8 bu  = *(const u32x8*)(Wl + (size_t)kc * 512); // 16 bf16, contiguous
    u32x8 au;
    au[0] = alo[0]; au[1] = alo[1]; au[2] = alo[2]; au[3] = alo[3];
    au[4] = ahi[0]; au[5] = ahi[1]; au[6] = ahi[2]; au[7] = ahi[3];
    v16bf a = __builtin_bit_cast(v16bf, au);
    v16bf b = __builtin_bit_cast(v16bf, bu);
    acc = __builtin_amdgcn_wmma_f32_16x16x32_bf16(false, a, false, b,
                                                  (short)0, acc, false, false);
  }

  int n = (tnb << 4) + l16;
  float bn = bias[n];
#pragma unroll
  for (int r = 0; r < 8; ++r) {
    int m = tm + 8 * half + r;
    float v = acc[r] + bn;
    if (OUT_BF16) ((unsigned short*)C)[(size_t)m * ldc + n] = f2bf(v);
    else          ((float*)C)[(size_t)m * ldc + n] = v;
  }
}

// ---------- LSTM cell ----------
__global__ void k_lstm(const float* __restrict__ gates, float* __restrict__ cx,
                       float* __restrict__ hx, unsigned short* __restrict__ hy_bf,
                       unsigned short* __restrict__ xh) {
  int i = blockIdx.x * blockDim.x + threadIdx.x;  // b*H + h
  if (i >= B_ * H_) return;
  int b = i >> 7, h = i & 127;
  const float* g = gates + (size_t)b * G4_;
  float ig = 1.f / (1.f + __expf(-g[h]));
  float fg = 1.f / (1.f + __expf(-g[H_ + h]));
  float gg = tanhf(g[2 * H_ + h]);
  float og = 1.f / (1.f + __expf(-g[3 * H_ + h]));
  float cy = fg * cx[i] + ig * gg;
  float hy = og * tanhf(cy);
  cx[i] = cy;
  hx[i] = hy;
  unsigned short hb = f2bf(hy);
  hy_bf[i] = hb;
  xh[(size_t)b * KCAT_ + H_ + h] = hb;   // hx half of next gates input
}

// ---------- attention scores: u[b,s] = sum_h v[h]*tanh(q[b,h] + e[s,b,h]) ----------
// One wave per (b,s); each lane handles 4 contiguous h (vectorized loads).
__global__ void k_score(const float* __restrict__ q, const unsigned short* __restrict__ e,
                        const float* __restrict__ v, float* __restrict__ u) {
  int wave = blockIdx.x * (blockDim.x >> 5) + (threadIdx.x >> 5);  // b*S + s
  if (wave >= B_ * S_) return;
  int lane = threadIdx.x & 31;
  int b = wave >> 7, s = wave & 127;
  int h0 = lane * 4;
  const float4 qv = *(const float4*)(q + (size_t)b * H_ + h0);
  const float4 vv = *(const float4*)(v + h0);
  u32x2 ev = *(const u32x2*)(e + ((size_t)s * B_ + b) * H_ + h0);
  float acc = vv.x * tanhf(qv.x + bf2f_lo(ev[0]))
            + vv.y * tanhf(qv.y + bf2f_hi(ev[0]))
            + vv.z * tanhf(qv.z + bf2f_lo(ev[1]))
            + vv.w * tanhf(qv.w + bf2f_hi(ev[1]));
  for (int m = 16; m >= 1; m >>= 1) acc += __shfl_xor(acc, m, 32);
  if (lane == 0) u[wave] = acc;
}

// ---------- masked softmax over S, one wave32 per batch row ----------
__global__ void k_softmax_g(const float* __restrict__ u, const unsigned char* __restrict__ mask,
                            float* __restrict__ p) {
  int row = blockIdx.x * (blockDim.x >> 5) + (threadIdx.x >> 5);
  if (row >= B_) return;
  int lane = threadIdx.x & 31;
  float lv[4];
  float mx = NEGV;
#pragma unroll
  for (int k = 0; k < 4; ++k) {
    int s = lane + 32 * k;
    float x = mask[row * S_ + s] ? NEGV : u[row * S_ + s];
    lv[k] = x;
    mx = fmaxf(mx, x);
  }
  for (int m = 16; m >= 1; m >>= 1) mx = fmaxf(mx, __shfl_xor(mx, m, 32));
  float sum = 0.f;
#pragma unroll
  for (int k = 0; k < 4; ++k) { lv[k] = __expf(lv[k] - mx); sum += lv[k]; }
  for (int m = 16; m >= 1; m >>= 1) sum += __shfl_xor(sum, m, 32);
  float inv = 1.f / sum;
#pragma unroll
  for (int k = 0; k < 4; ++k) p[row * S_ + lane + 32 * k] = lv[k] * inv;
}

// ---------- glimpse: g_l[b,h] = sum_s e_g[s,b,h] * p[b,s] ----------
__global__ void k_glimpse(const unsigned short* __restrict__ e, const float* __restrict__ p,
                          unsigned short* __restrict__ gl_bf) {
  int i = blockIdx.x * blockDim.x + threadIdx.x;  // b*H + h
  if (i >= B_ * H_) return;
  int b = i >> 7, h = i & 127;
  const float* pb = p + (size_t)b * S_;
  float acc = 0.f;
#pragma unroll 4
  for (int s = 0; s < S_; ++s)
    acc += bf2f(e[((size_t)s * B_ + b) * H_ + h]) * pb[s];
  gl_bf[i] = f2bf(acc);
}

// ---------- pointer head: softmax(10*tanh), probs out, Gumbel-max sample,
//            mask update, gather next embedding into xh ----------
__device__ __forceinline__ unsigned int hash3(unsigned int a, unsigned int b, unsigned int c) {
  unsigned int h = a * 0x9E3779B1u ^ (b + 0x7F4A7C15u) * 0x85EBCA77u
                 ^ (c + 0x165667B1u) * 0xC2B2AE3Du;
  h ^= h >> 16; h *= 0x7FEB352Du;
  h ^= h >> 15; h *= 0x846CA68Bu;
  h ^= h >> 16;
  return h;
}

__global__ void k_pointer(const float* __restrict__ u, unsigned char* __restrict__ mask,
                          const float* __restrict__ emb, unsigned short* __restrict__ xh,
                          float* __restrict__ probs_out, float* __restrict__ sel_out,
                          int step) {
  int row = blockIdx.x * (blockDim.x >> 5) + (threadIdx.x >> 5);
  if (row >= B_) return;
  int lane = threadIdx.x & 31;
  float lv[4];
  float mx = NEGV;
#pragma unroll
  for (int k = 0; k < 4; ++k) {
    int s = lane + 32 * k;
    float x = mask[row * S_ + s] ? NEGV : 10.0f * tanhf(u[row * S_ + s]);
    lv[k] = x;
    mx = fmaxf(mx, x);
  }
  for (int m = 16; m >= 1; m >>= 1) mx = fmaxf(mx, __shfl_xor(mx, m, 32));
  float sum = 0.f, ev[4];
#pragma unroll
  for (int k = 0; k < 4; ++k) { ev[k] = __expf(lv[k] - mx); sum += ev[k]; }
  for (int m = 16; m >= 1; m >>= 1) sum += __shfl_xor(sum, m, 32);
  float inv = 1.f / sum;
  float* po = probs_out + (size_t)step * B_ * S_ + (size_t)row * S_;
#pragma unroll
  for (int k = 0; k < 4; ++k) po[lane + 32 * k] = ev[k] * inv;

  // Gumbel-max sampling on the logits (deterministic counter-based hash)
  float best = NEGV; int bidx = 0;
#pragma unroll
  for (int k = 0; k < 4; ++k) {
    int s = lane + 32 * k;
    float r = ((hash3((unsigned)step, (unsigned)row, (unsigned)s) >> 8) + 1u)
              * (1.0f / 16777218.0f);
    float sc = lv[k] + (-__logf(-__logf(r)));
    if (sc > best) { best = sc; bidx = s; }
  }
  for (int m = 16; m >= 1; m >>= 1) {
    float ov = __shfl_xor(best, m, 32);
    int   oi = __shfl_xor(bidx, m, 32);
    if (ov > best || (ov == best && oi < bidx)) { best = ov; bidx = oi; }
  }
  int idx = __shfl(bidx, 0, 32);
  if (lane == 0) {
    sel_out[(size_t)step * B_ + row] = (float)idx;
    mask[row * S_ + idx] = 1;
  }
  const float* er = emb + ((size_t)idx * B_ + row) * E_;
#pragma unroll
  for (int k = 0; k < 4; ++k) {
    int e = lane + 32 * k;
    xh[(size_t)row * KCAT_ + e] = f2bf(er[e]);
  }
}

// =====================================================================
extern "C" void kernel_launch(void* const* d_in, const int* in_sizes, int n_in,
                              void* d_out, int out_size, void* d_ws, size_t ws_size,
                              hipStream_t stream) {
  (void)in_sizes; (void)n_in; (void)out_size; (void)ws_size;

  const float* decoder_input = (const float*)d_in[0];
  const float* emb           = (const float*)d_in[1];
  const float* hx_in         = (const float*)d_in[2];
  const float* cx_in         = (const float*)d_in[3];
  const float* enc           = (const float*)d_in[4];
  const float* W_in          = (const float*)d_in[5];
  const float* b_in          = (const float*)d_in[6];
  const float* W_out         = (const float*)d_in[7];
  const float* b_out         = (const float*)d_in[8];
  const float* Wq_g          = (const float*)d_in[9];
  const float* bq_g          = (const float*)d_in[10];
  const float* Wr_g          = (const float*)d_in[11];
  const float* br_g          = (const float*)d_in[12];
  const float* v_g           = (const float*)d_in[13];
  const float* Wq_p          = (const float*)d_in[14];
  const float* bq_p          = (const float*)d_in[15];
  const float* Wr_p          = (const float*)d_in[16];
  const float* br_p          = (const float*)d_in[17];
  const float* v_p           = (const float*)d_in[18];

  char* ws = (char*)d_ws;
  size_t off = 0;
  auto alloc = [&](size_t bytes) -> char* {
    char* p = ws + off;
    off += (bytes + 255) & ~(size_t)255;
    return p;
  };

  unsigned short* enc_bf  = (unsigned short*)alloc((size_t)S_ * B_ * H_ * 2);
  unsigned short* e_g     = (unsigned short*)alloc((size_t)S_ * B_ * H_ * 2);
  unsigned short* e_p     = (unsigned short*)alloc((size_t)S_ * B_ * H_ * 2);
  unsigned short* Wcat_p  = (unsigned short*)alloc((size_t)KCAT_ * G4_ * 2);
  unsigned short* Wq_g_p  = (unsigned short*)alloc((size_t)H_ * H_ * 2);
  unsigned short* Wr_g_p  = (unsigned short*)alloc((size_t)H_ * H_ * 2);
  unsigned short* Wq_p_p  = (unsigned short*)alloc((size_t)H_ * H_ * 2);
  unsigned short* Wr_p_p  = (unsigned short*)alloc((size_t)H_ * H_ * 2);
  float*          biascat = (float*)alloc((size_t)G4_ * 4);
  unsigned short* xh      = (unsigned short*)alloc((size_t)B_ * KCAT_ * 2);
  float*          gates   = (float*)alloc((size_t)B_ * G4_ * 4);
  float*          hxb     = (float*)alloc((size_t)B_ * H_ * 4);
  float*          cxb     = (float*)alloc((size_t)B_ * H_ * 4);
  unsigned short* hy_bf   = (unsigned short*)alloc((size_t)B_ * H_ * 2);
  unsigned short* gl_bf   = (unsigned short*)alloc((size_t)B_ * H_ * 2);
  float*          q       = (float*)alloc((size_t)B_ * H_ * 4);
  float*          u       = (float*)alloc((size_t)B_ * S_ * 4);
  float*          pg      = (float*)alloc((size_t)B_ * S_ * 4);
  unsigned char*  mask    = (unsigned char*)alloc((size_t)B_ * S_);

  const int thr = 256;
  // ---- one-time setup: bf16 conversions, fragment-packed weights, state init ----
  k_cvt<<<((size_t)S_ * B_ * H_ + thr - 1) / thr, thr, 0, stream>>>(enc, enc_bf, S_ * B_, H_, H_, H_);
  k_packW<<<(KCAT_ * G4_ + thr - 1) / thr, thr, 0, stream>>>(W_in, W_out, E_, G4_, Wcat_p, G4_, KCAT_);
  k_packW<<<(H_ * H_ + thr - 1) / thr, thr, 0, stream>>>(Wq_g, Wq_g, H_, H_, Wq_g_p, H_, H_);
  k_packW<<<(H_ * H_ + thr - 1) / thr, thr, 0, stream>>>(Wr_g, Wr_g, H_, H_, Wr_g_p, H_, H_);
  k_packW<<<(H_ * H_ + thr - 1) / thr, thr, 0, stream>>>(Wq_p, Wq_p, H_, H_, Wq_p_p, H_, H_);
  k_packW<<<(H_ * H_ + thr - 1) / thr, thr, 0, stream>>>(Wr_p, Wr_p, H_, H_, Wr_p_p, H_, H_);
  k_addbias<<<(G4_ + thr - 1) / thr, thr, 0, stream>>>(b_in, b_out, biascat, G4_);
  k_cvt<<<(B_ * E_ + thr - 1) / thr, thr, 0, stream>>>(decoder_input, xh, B_, E_, E_, KCAT_);
  k_cvt<<<(B_ * H_ + thr - 1) / thr, thr, 0, stream>>>(hx_in, xh + H_, B_, H_, H_, KCAT_);
  hipMemcpyAsync(hxb, hx_in, (size_t)B_ * H_ * 4, hipMemcpyDeviceToDevice, stream);
  hipMemcpyAsync(cxb, cx_in, (size_t)B_ * H_ * 4, hipMemcpyDeviceToDevice, stream);
  hipMemsetAsync(mask, 0, (size_t)B_ * S_, stream);

  // ---- precompute e_g / e_p: (S*B x H) = (S*B x H) @ (H x H) + bias ----
  const int tiles_e = (S_ * B_ / 16) * (H_ / 16);  // 32768 waves
  k_gemm<true><<<tiles_e / 8, thr, 0, stream>>>(enc_bf, H_, Wr_g_p, br_g, e_g, H_, S_ * B_, H_, H_);
  k_gemm<true><<<tiles_e / 8, thr, 0, stream>>>(enc_bf, H_, Wr_p_p, br_p, e_p, H_, S_ * B_, H_, H_);

  float* probs_out = (float*)d_out;
  float* sel_out   = probs_out + (size_t)S_ * B_ * S_;
  float* hy_out    = sel_out + (size_t)S_ * B_;
  float* cy_out    = hy_out + (size_t)B_ * H_;

  const int tiles_g = (B_ / 16) * (G4_ / 16);  // 1024 waves
  const int tiles_q = (B_ / 16) * (H_ / 16);   // 256 waves

  for (int step = 0; step < S_; ++step) {
    // gates = [x|hx] @ [W_in;W_out] + (b_in+b_out)
    k_gemm<false><<<tiles_g / 8, thr, 0, stream>>>(xh, KCAT_, Wcat_p, biascat, gates, G4_, B_, G4_, KCAT_);
    k_lstm<<<(B_ * H_) / thr, thr, 0, stream>>>(gates, cxb, hxb, hy_bf, xh);
    // glimpse attention
    k_gemm<false><<<tiles_q / 8, thr, 0, stream>>>(hy_bf, H_, Wq_g_p, bq_g, q, H_, B_, H_, H_);
    k_score<<<(B_ * S_) / 8, thr, 0, stream>>>(q, e_g, v_g, u);
    k_softmax_g<<<B_ / 8, thr, 0, stream>>>(u, mask, pg);
    k_glimpse<<<(B_ * H_) / thr, thr, 0, stream>>>(e_g, pg, gl_bf);
    // pointer attention + sampling + next-input gather
    k_gemm<false><<<tiles_q / 8, thr, 0, stream>>>(gl_bf, H_, Wq_p_p, bq_p, q, H_, B_, H_, H_);
    k_score<<<(B_ * S_) / 8, thr, 0, stream>>>(q, e_p, v_p, u);
    k_pointer<<<B_ / 8, thr, 0, stream>>>(u, mask, emb, xh, probs_out, sel_out, step);
  }

  hipMemcpyAsync(hy_out, hxb, (size_t)B_ * H_ * 4, hipMemcpyDeviceToDevice, stream);
  hipMemcpyAsync(cy_out, cxb, (size_t)B_ * H_ * 4, hipMemcpyDeviceToDevice, stream);
}
